// GraphMSG_5488968204435
// MI455X (gfx1250) — compile-verified, matrix operand order checked by reading, not run on hardware
//
#include <hip/hip_runtime.h>
#include <cstdint>

// ---------------------------------------------------------------------------
// GraphMSG (AIFS-like GNN) for MI455X / gfx1250.
// All GEMMs run through v_wmma_f32_16x16x32_bf16 (bf16 inputs, f32 accum).
// Each MLP (Linear->SiLU->Linear->LayerNorm [+residual]) is one fused kernel.
// Block = 128 threads (4 waves) x 64 rows (4 row-tiles): each weight fragment
// feeds 4 WMMAs (8 wmma : 4 global_load_b128 per K-step per wave).
// ---------------------------------------------------------------------------

typedef __bf16 bf16;
typedef __attribute__((ext_vector_type(8)))  bf16  v8bf;
typedef __attribute__((ext_vector_type(16))) bf16  v16bf;
typedef __attribute__((ext_vector_type(8)))  float v8f;

#define ERA_N   40320
#define H_N     10242
#define NE_E2H  161280
#define NE_H2H  61440
#define NE_H2E  120960
#define NF      98          // IN_CH + AUX_CH
#define IN_CH   80
#define HID     128
#define ROWS    64          // rows per block (4 row-tiles of 16)
#define XS_MAX  392         // max din_pad (384) + 8 pad

// Load a 16-element bf16 WMMA fragment: per CDNA5 ISA, a lane's 16 values are
// two contiguous runs of 8 (K = base..base+7 and base+16..base+23).
__device__ __forceinline__ v16bf ld16(const bf16* p) {
  v8bf lo = *(const v8bf*)p;
  v8bf hi = *(const v8bf*)(p + 16);
  v16bf r;
#pragma unroll
  for (int i = 0; i < 8; ++i) { r[i] = lo[i]; r[i + 8] = hi[i]; }
  return r;
}

// ---------------------------------------------------------------------------
// Fused MLP: out = LayerNorm(W2^T(SiLU(W1^T x + b1)) + b2)*g + bt [+ res]
// x = concat of up to 3 row-gathered parts. One block = 64 rows, 4 waves;
// each wave owns 32 output columns and 4 row-tiles (8 WMMA accumulators).
// ---------------------------------------------------------------------------
__global__ void __launch_bounds__(128)
mlp_ln_kernel(const float* __restrict__ x0, int d0, const int* __restrict__ idx0,
              const float* __restrict__ x1, int d1, const int* __restrict__ idx1,
              const float* __restrict__ x2, int d2, const int* __restrict__ idx2,
              const bf16* __restrict__ w1t, const float* __restrict__ b1,
              const bf16* __restrict__ w2t, const float* __restrict__ b2,
              const float* __restrict__ gam, const float* __restrict__ bet,
              const float* __restrict__ res,
              float* __restrict__ out, int nrows, int din_pad) {
  __shared__ bf16  sX[ROWS * XS_MAX]; // input tile, bf16, padded stride
  __shared__ bf16  sH[ROWS * 136];    // hidden tile (post-SiLU)
  __shared__ float sO[ROWS * 132];    // pre-LN output tile
  __shared__ float sMu[ROWS];
  __shared__ float sRs[ROWS];

  const int tid  = threadIdx.x;
  const int lane = tid & 31;
  const int wave = tid >> 5;
  const int row0 = blockIdx.x * ROWS;
  const int din  = d0 + d1 + d2;
  const int XS   = din_pad + 8;       // keeps 16B alignment per row

  // ---- stage X tile (concat + optional gather) into LDS as bf16 ----
  for (int i = tid; i < ROWS * XS; i += 128) {
    const int r = i / XS, c = i - r * XS;
    const int gr = row0 + r;
    float v = 0.0f;
    if (gr < nrows && c < din) {
      if (c < d0) {
        const int rr = idx0 ? idx0[gr] : gr;
        v = x0[(long)rr * d0 + c];
      } else if (c < d0 + d1) {
        const int rr = idx1 ? idx1[gr] : gr;
        v = x1[(long)rr * d1 + (c - d0)];
      } else {
        const int rr = idx2 ? idx2[gr] : gr;
        v = x2[(long)rr * d2 + (c - d0 - d1)];
      }
    }
    sX[i] = (bf16)v;
  }
  __syncthreads();

  const int nlo   = lane & 15;          // A row / B,C,D column within tile
  const int khalf = (lane >> 4) * 8;    // K-half select for this lane
  const int col0  = wave * 32 + nlo;
  const int col1  = col0 + 16;

  // ---- GEMM1: h = x @ W1 + b1 ----
  v8f acc[4][2];
  {
    const float bb0 = b1[col0], bb1 = b1[col1];
#pragma unroll
    for (int rt = 0; rt < 4; ++rt)
#pragma unroll
      for (int r = 0; r < 8; ++r) { acc[rt][0][r] = bb0; acc[rt][1][r] = bb1; }
  }
  const int nk1 = din_pad >> 5;
  for (int kc = 0; kc < nk1; ++kc) {
    const int k0 = kc * 32 + khalf;
    const v16bf bA = ld16(&w1t[(long)col0 * din_pad + k0]);
    const v16bf bB = ld16(&w1t[(long)col1 * din_pad + k0]);
#pragma unroll
    for (int rt = 0; rt < 4; ++rt) {
      const v16bf a = ld16(&sX[(rt * 16 + nlo) * XS + k0]);
      acc[rt][0] = __builtin_amdgcn_wmma_f32_16x16x32_bf16(false, a, false, bA, (short)0, acc[rt][0], false, false);
      acc[rt][1] = __builtin_amdgcn_wmma_f32_16x16x32_bf16(false, a, false, bB, (short)0, acc[rt][1], false, false);
    }
  }

  // ---- SiLU, stage hidden tile as bf16 ----
#pragma unroll
  for (int rt = 0; rt < 4; ++rt)
#pragma unroll
    for (int r = 0; r < 8; ++r) {
      const int m = rt * 16 + r + khalf;
      float v0 = acc[rt][0][r]; v0 = v0 / (1.0f + __expf(-v0));
      float v1 = acc[rt][1][r]; v1 = v1 / (1.0f + __expf(-v1));
      sH[m * 136 + col0] = (bf16)v0;
      sH[m * 136 + col1] = (bf16)v1;
    }
  __syncthreads();

  // ---- GEMM2: o = h @ W2 + b2 ----
  v8f c2[4][2];
  {
    const float bb0 = b2[col0], bb1 = b2[col1];
#pragma unroll
    for (int rt = 0; rt < 4; ++rt)
#pragma unroll
      for (int r = 0; r < 8; ++r) { c2[rt][0][r] = bb0; c2[rt][1][r] = bb1; }
  }
#pragma unroll
  for (int kc = 0; kc < 4; ++kc) {
    const int k0 = kc * 32 + khalf;
    const v16bf bA = ld16(&w2t[(long)col0 * HID + k0]);
    const v16bf bB = ld16(&w2t[(long)col1 * HID + k0]);
#pragma unroll
    for (int rt = 0; rt < 4; ++rt) {
      const v16bf a = ld16(&sH[(rt * 16 + nlo) * 136 + k0]);
      c2[rt][0] = __builtin_amdgcn_wmma_f32_16x16x32_bf16(false, a, false, bA, (short)0, c2[rt][0], false, false);
      c2[rt][1] = __builtin_amdgcn_wmma_f32_16x16x32_bf16(false, a, false, bB, (short)0, c2[rt][1], false, false);
    }
  }
#pragma unroll
  for (int rt = 0; rt < 4; ++rt)
#pragma unroll
    for (int r = 0; r < 8; ++r) {
      const int m = rt * 16 + r + khalf;
      sO[m * 132 + col0] = c2[rt][0][r];
      sO[m * 132 + col1] = c2[rt][1][r];
    }
  __syncthreads();

  // ---- LayerNorm over 128 features per row ----
  if (tid < ROWS) {
    float mu = 0.0f;
    for (int c = 0; c < HID; ++c) mu += sO[tid * 132 + c];
    mu *= (1.0f / 128.0f);
    float var = 0.0f;
    for (int c = 0; c < HID; ++c) { const float d = sO[tid * 132 + c] - mu; var += d * d; }
    var *= (1.0f / 128.0f);
    sMu[tid] = mu;
    sRs[tid] = __frsqrt_rn(var + 1e-5f);
  }
  __syncthreads();

  for (int i = tid; i < ROWS * HID; i += 128) {
    const int r = i >> 7, c = i & 127;
    const int gr = row0 + r;
    if (gr < nrows) {
      float v = (sO[r * 132 + c] - sMu[r]) * sRs[r] * gam[c] + bet[c];
      if (res) v += res[(long)gr * HID + c];
      out[(long)gr * HID + c] = v;
    }
  }
}

// ---------------------------------------------------------------------------
// Weight prep: f32 [din][dout] -> bf16 transposed+padded [dout][din_pad]
// ---------------------------------------------------------------------------
__global__ void convert_wt_kernel(const float* __restrict__ w, bf16* __restrict__ wt,
                                  int din, int din_pad, int dout) {
  const long total = (long)dout * din_pad;
  for (long i = (long)blockIdx.x * blockDim.x + threadIdx.x; i < total;
       i += (long)gridDim.x * blockDim.x) {
    const int n = (int)(i / din_pad), k = (int)(i - (long)n * din_pad);
    wt[i] = (k < din) ? (bf16)w[(long)k * dout + n] : (bf16)0.0f;
  }
}

// x_era = concat('b m n f -> (b n) (m f)', era_latlons, era_trainable) : [ERA, 208]
__global__ void build_xera_kernel(const float* __restrict__ x, const float* __restrict__ ll,
                                  const float* __restrict__ tr, float* __restrict__ out) {
  const long total = (long)ERA_N * 208;
  for (long i = (long)blockIdx.x * blockDim.x + threadIdx.x; i < total;
       i += (long)gridDim.x * blockDim.x) {
    const int n = (int)(i / 208), c = (int)(i - (long)n * 208);
    float v;
    if (c < 2 * NF)      { const int m = c / NF, f = c - m * NF; v = x[((long)m * ERA_N + n) * NF + f]; }
    else if (c < 2 * NF + 4) v = ll[(long)n * 4 + (c - 2 * NF)];
    else                     v = tr[(long)n * 8 + (c - 2 * NF - 4)];
    out[i] = v;
  }
}

__global__ void cat2_kernel(const float* __restrict__ a, int wa,
                            const float* __restrict__ b, int wb,
                            float* __restrict__ out, int n) {
  const int w = wa + wb;
  const long total = (long)n * w;
  for (long i = (long)blockIdx.x * blockDim.x + threadIdx.x; i < total;
       i += (long)gridDim.x * blockDim.x) {
    const int r = (int)(i / w), c = (int)(i - (long)r * w);
    out[i] = (c < wa) ? a[(long)r * wa + c] : b[(long)r * wb + (c - wa)];
  }
}

__global__ void zero_kernel(float* __restrict__ p, long n) {
  for (long i = (long)blockIdx.x * blockDim.x + threadIdx.x; i < n;
       i += (long)gridDim.x * blockDim.x) p[i] = 0.0f;
}

__global__ void scatter_add_kernel(const float* __restrict__ m, const int* __restrict__ dst,
                                   float* __restrict__ agg, int ne) {
  const long total = (long)ne * HID;
  for (long i = (long)blockIdx.x * blockDim.x + threadIdx.x; i < total;
       i += (long)gridDim.x * blockDim.x) {
    const int e = (int)(i >> 7), c = (int)(i & 127);
    atomicAdd(&agg[(long)dst[e] * HID + c], m[i]);
  }
}

__global__ void add_kernel(const float* __restrict__ a, const float* __restrict__ b,
                           float* __restrict__ o, long n) {
  for (long i = (long)blockIdx.x * blockDim.x + threadIdx.x; i < n;
       i += (long)gridDim.x * blockDim.x) o[i] = a[i] + b[i];
}

// out[n,c] = xd[n]@W[:,c] + b[c] + x_last[n,c]   (c < 80) -- tiny tail GEMM
__global__ void __launch_bounds__(128)
final_out_kernel(const float* __restrict__ xd, const float* __restrict__ w,
                 const float* __restrict__ b, const float* __restrict__ xlast,
                 float* __restrict__ out) {
  __shared__ float sx[HID];
  const int n = blockIdx.x;
  sx[threadIdx.x] = xd[(long)n * HID + threadIdx.x];
  __syncthreads();
  const int c = threadIdx.x;
  if (c < IN_CH) {
    float acc = b[c];
#pragma unroll 8
    for (int k = 0; k < HID; ++k) acc += sx[k] * w[(long)k * IN_CH + c];
    out[(long)n * IN_CH + c] = acc + xlast[(long)n * NF + c];
  }
}

// ---------------------------------------------------------------------------
extern "C" void kernel_launch(void* const* d_in, const int* in_sizes, int n_in,
                              void* d_out, int out_size, void* d_ws, size_t ws_size,
                              hipStream_t stream) {
  (void)in_sizes; (void)n_in; (void)out_size; (void)ws_size;

  int p = 0;
  auto F = [&]() { return (const float*)d_in[p++]; };

  // ---- inputs in setup_inputs() insertion order (params leaves recursively) ----
  const float* x = F();                         // [1,2,ERA,98]
  const float* era_tr = F();
  const float* h_tr   = F();
  const float* e2h_tr = F();
  const float* h2h_tr = F();
  const float* h2e_tr = F();
  struct MlpP { const float *w1, *b1, *w2, *b2, *g, *bt; };
  auto getMlp = [&]() { MlpP m; m.w1 = F(); m.b1 = F(); m.w2 = F(); m.b2 = F(); m.g = F(); m.bt = F(); return m; };
  MlpP fm_src = getMlp(), fm_dst = getMlp(), fm_edge = getMlp();
  MlpP fm_blk_e = getMlp(), fm_blk_n = getMlp();
  MlpP proc_edge = getMlp();
  MlpP proc_e[4], proc_n[4];
  for (int i = 0; i < 4; ++i) { proc_e[i] = getMlp(); proc_n[i] = getMlp(); }
  MlpP bm_src = getMlp(), bm_dst = getMlp(), bm_edge = getMlp();
  MlpP bm_blk_e = getMlp(), bm_blk_n = getMlp();
  const float* bm_out_w = F();
  const float* bm_out_b = F();
  const float* era_latlons = F();
  const float* h_latlons   = F();
  const float* e2h_attr = F();
  const float* h2h_attr = F();
  const float* h2e_attr = F();
  const int* e2h_idx = (const int*)d_in[p++];   // [2, E]
  const int* h2h_idx = (const int*)d_in[p++];
  const int* h2e_idx = (const int*)d_in[p++];
  const int *e2h_s = e2h_idx, *e2h_d = e2h_idx + NE_E2H;
  const int *h2h_s = h2h_idx, *h2h_d = h2h_idx + NE_H2H;
  const int *h2e_s = h2e_idx, *h2e_d = h2e_idx + NE_H2E;

  // ---- workspace bump allocator ----
  char* wsp = (char*)d_ws;
  auto allocF = [&](size_t nf) { float* r = (float*)wsp; wsp += ((nf * 4 + 255) / 256) * 256; return r; };
  auto allocH = [&](size_t nh) { bf16*  r = (bf16*)wsp;  wsp += ((nh * 2 + 255) / 256) * 256; return r; };

  float* xera_in  = allocF((size_t)ERA_N * 208);
  float* xh_in    = allocF((size_t)H_N * 12);
  float* e2h_feat = allocF((size_t)NE_E2H * 11);
  float* h2h_feat = allocF((size_t)NE_H2H * 11);
  float* h2e_feat = allocF((size_t)NE_H2E * 11);
  float* x_era_lat = allocF((size_t)ERA_N * HID);
  float* x_latent  = allocF((size_t)H_N * HID);
  float* xp        = allocF((size_t)H_N * HID);
  float* xs_buf    = allocF((size_t)H_N * HID);
  float* xd_buf    = allocF((size_t)ERA_N * HID);
  float* ebuf      = allocF((size_t)NE_E2H * HID);   // reused per phase (max E)
  float* agg       = allocF((size_t)ERA_N * HID);    // max(n_dst) * HID

  // ---- weight prep (f32 -> transposed, zero-padded bf16) ----
  struct MlpW { const bf16 *w1t, *w2t; const float *b1, *b2, *g, *bt; int din_pad; };
  auto prep = [&](const MlpP& m, int din) {
    MlpW w;
    w.din_pad = ((din + 31) / 32) * 32;
    bf16* w1t = allocH((size_t)HID * w.din_pad);
    bf16* w2t = allocH((size_t)HID * HID);
    convert_wt_kernel<<<512, 256, 0, stream>>>(m.w1, w1t, din, w.din_pad, HID);
    convert_wt_kernel<<<256, 256, 0, stream>>>(m.w2, w2t, HID, HID, HID);
    w.w1t = w1t; w.w2t = w2t; w.b1 = m.b1; w.b2 = m.b2; w.g = m.g; w.bt = m.bt;
    return w;
  };
  MlpW W_fm_src = prep(fm_src, 208), W_fm_dst = prep(fm_dst, 12), W_fm_edge = prep(fm_edge, 11);
  MlpW W_fm_be = prep(fm_blk_e, 384), W_fm_bn = prep(fm_blk_n, 256);
  MlpW W_pe = prep(proc_edge, 11);
  MlpW W_pbe[4], W_pbn[4];
  for (int i = 0; i < 4; ++i) { W_pbe[i] = prep(proc_e[i], 384); W_pbn[i] = prep(proc_n[i], 256); }
  MlpW W_bm_src = prep(bm_src, 128), W_bm_dst = prep(bm_dst, 128), W_bm_edge = prep(bm_edge, 11);
  MlpW W_bm_be = prep(bm_blk_e, 384), W_bm_bn = prep(bm_blk_n, 256);

  auto mlp = [&](const MlpW& W,
                 const float* a0, int d0, const int* i0,
                 const float* a1, int d1, const int* i1,
                 const float* a2, int d2, const int* i2,
                 const float* res, float* out, int nrows) {
    mlp_ln_kernel<<<(nrows + ROWS - 1) / ROWS, 128, 0, stream>>>(
        a0, d0, i0, a1, d1, i1, a2, d2, i2,
        W.w1t, W.b1, W.w2t, W.b2, W.g, W.bt, res, out, nrows, W.din_pad);
  };

  // ---- assemble raw inputs ----
  build_xera_kernel<<<2048, 256, 0, stream>>>(x, era_latlons, era_tr, xera_in);
  cat2_kernel<<<256, 256, 0, stream>>>(h_latlons, 4, h_tr, 8, xh_in, H_N);
  cat2_kernel<<<1024, 256, 0, stream>>>(e2h_attr, 3, e2h_tr, 8, e2h_feat, NE_E2H);
  cat2_kernel<<<512, 256, 0, stream>>>(h2h_attr, 3, h2h_tr, 8, h2h_feat, NE_H2H);
  cat2_kernel<<<1024, 256, 0, stream>>>(h2e_attr, 3, h2e_tr, 8, h2e_feat, NE_H2E);

  // ---- forward mapper (era -> h) ----
  mlp(W_fm_edge, e2h_feat, 11, 0, 0, 0, 0, 0, 0, 0, 0, ebuf, NE_E2H);
  mlp(W_fm_src, xera_in, 208, 0, 0, 0, 0, 0, 0, 0, 0, x_era_lat, ERA_N);
  mlp(W_fm_dst, xh_in, 12, 0, 0, 0, 0, 0, 0, 0, 0, x_latent, H_N);
  mlp(W_fm_be, x_era_lat, HID, e2h_s, x_latent, HID, e2h_d, ebuf, HID, 0, ebuf, ebuf, NE_E2H);
  zero_kernel<<<1024, 256, 0, stream>>>(agg, (long)H_N * HID);
  scatter_add_kernel<<<4096, 256, 0, stream>>>(ebuf, e2h_d, agg, NE_E2H);
  mlp(W_fm_bn, x_latent, HID, 0, agg, HID, 0, 0, 0, 0, x_latent, x_latent, H_N);

  // ---- hidden processor (h -> h), 4 layers ----
  hipMemcpyAsync(xp, x_latent, (size_t)H_N * HID * sizeof(float),
                 hipMemcpyDeviceToDevice, stream);
  mlp(W_pe, h2h_feat, 11, 0, 0, 0, 0, 0, 0, 0, 0, ebuf, NE_H2H);
  for (int l = 0; l < 4; ++l) {
    mlp(W_pbe[l], xp, HID, h2h_s, xp, HID, h2h_d, ebuf, HID, 0, ebuf, ebuf, NE_H2H);
    zero_kernel<<<1024, 256, 0, stream>>>(agg, (long)H_N * HID);
    scatter_add_kernel<<<2048, 256, 0, stream>>>(ebuf, h2h_d, agg, NE_H2H);
    mlp(W_pbn[l], xp, HID, 0, agg, HID, 0, 0, 0, 0, xp, xp, H_N);
  }
  add_kernel<<<1024, 256, 0, stream>>>(xp, x_latent, xp, (long)H_N * HID);

  // ---- backward mapper (h -> era) ----
  mlp(W_bm_edge, h2e_feat, 11, 0, 0, 0, 0, 0, 0, 0, 0, ebuf, NE_H2E);
  mlp(W_bm_src, xp, HID, 0, 0, 0, 0, 0, 0, 0, 0, xs_buf, H_N);
  mlp(W_bm_dst, x_era_lat, HID, 0, 0, 0, 0, 0, 0, 0, 0, xd_buf, ERA_N);
  mlp(W_bm_be, xs_buf, HID, h2e_s, xd_buf, HID, h2e_d, ebuf, HID, 0, ebuf, ebuf, NE_H2E);
  zero_kernel<<<2048, 256, 0, stream>>>(agg, (long)ERA_N * HID);
  scatter_add_kernel<<<4096, 256, 0, stream>>>(ebuf, h2e_d, agg, NE_H2E);
  mlp(W_bm_bn, xd_buf, HID, 0, agg, HID, 0, 0, 0, 0, xd_buf, xd_buf, ERA_N);

  // ---- output head + residual on x[:, -1, :, :80] ----
  final_out_kernel<<<ERA_N, 128, 0, stream>>>(xd_buf, bm_out_w, bm_out_b,
                                              x + (long)ERA_N * NF, (float*)d_out);
}
